// MPNN_20151986553340
// MI455X (gfx1250) — matrix operand order repaired and checked by reading, not compiled, not run on hardware
//
#include <hip/hip_runtime.h>

#define NATOMS 500000
#define NEDGES 2000000
#define NMOLS  50000
#define FDIM   75
#define FP     80           // padded bf16 row length (160 B, 16B-aligned)
#define KPAD   160          // A = [h(80) | m(80)] -> 5 chunks of K=32
#define DEPTH  3

typedef __attribute__((ext_vector_type(16))) __bf16        v16bf;
typedef __attribute__((ext_vector_type(8)))  float         v8f;
typedef __attribute__((ext_vector_type(4)))  unsigned int  v4u;
typedef unsigned short ushort;

union ABfrag { v16bf bf; v4u q[2]; };

__device__ __forceinline__ ushort f2bf(float f) {
    unsigned u = __float_as_uint(f);
    u += 0x7FFFu + ((u >> 16) & 1u);          // round-to-nearest-even
    return (ushort)(u >> 16);
}
__device__ __forceinline__ float bf2f(ushort b) {
    return __uint_as_float(((unsigned)b) << 16);
}
__device__ __forceinline__ void atomic_fadd(float* p, float v) {
    __hip_atomic_fetch_add(p, v, __ATOMIC_RELAXED, __HIP_MEMORY_SCOPE_AGENT);
}

// ------------------------------------------------------------- zero (u32)
__global__ void zero_u32(unsigned* __restrict__ p, long n) {
    long i = (long)blockIdx.x * blockDim.x + threadIdx.x;
    if (i < n) p[i] = 0u;
}

// --------------------------- fp32 atom_feats -> bf16 rows padded to FP=80
__global__ void cvt_feats(const float* __restrict__ f, ushort* __restrict__ h) {
    long i = (long)blockIdx.x * blockDim.x + threadIdx.x;
    if (i >= (long)NATOMS * FP) return;
    int r = (int)(i / FP), c = (int)(i - (long)r * FP);
    h[i] = (c < FDIM) ? f2bf(f[(long)r * FDIM + c]) : (ushort)0;
}

// ---------------- edge scatter: m[dst] += h[src], packed bf16 atomics.
// One thread per (edge, feature-pair): 40 pairs/edge, one
// global_atomic_pk_add_bf16 per pair -> 80M atomics/layer (vs 150M fp32).
__global__ void edge_scatter(const ushort* __restrict__ h,
                             const int*   __restrict__ src,
                             const int*   __restrict__ dst,
                             ushort*      __restrict__ m) {
    long i = (long)blockIdx.x * blockDim.x + threadIdx.x;
    if (i >= (long)NEDGES * (FP / 2)) return;
    int e = (int)(i / (FP / 2));
    int p = (int)(i - (long)e * (FP / 2));
    unsigned pair = *(const unsigned*)(h + (long)src[e] * FP + 2 * p);
    unsigned voff = (unsigned)dst[e] * (FP * 2) + p * 4;   // byte offset, < 2^31
    asm volatile("global_atomic_pk_add_bf16 %0, %1, %2"
                 :: "v"(voff), "v"(pair), "s"((const void*)m) : "memory");
}

// ------------------- fused concat + GEMM + bias + ReLU (bf16 WMMA)
// out[r, j] = relu( sum_k concat(h,m)[r,k] * W[j,k] + b[j] ), bf16 out.
// Block: 128 threads = 4 wave32, 64 rows; per wave a 16x80 tile, 5 K32 chunks.
// A tile staged by CDNA5 async copies (global_load_async_to_lds_b128).
__global__ void __launch_bounds__(128)
mpnn_gemm(const ushort* __restrict__ h, const ushort* __restrict__ m,
          const float* __restrict__ W, const float* __restrict__ bias,
          ushort* __restrict__ out) {
    __shared__ ushort ldsA[64 * KPAD];        // 20480 B: [h(80)|m(80)] rows
    __shared__ ushort ldsB[5 * 5 * 32 * 16];  // 25600 B, fragment-major

    const int tid  = threadIdx.x;
    const int lane = tid & 31;
    const int wave = tid >> 5;
    const int rl   = lane & 15;
    const int hi   = (lane >> 4) & 1;
    const long rowBase = (long)blockIdx.x * 64;

    // Async-copy A tile: 64 rows x (160B from h + 160B from m) = 1280 B128 ops.
    const unsigned ldsAoff = (unsigned)(uintptr_t)(&ldsA[0]);
    for (int s = tid; s < 64 * 20; s += 128) {
        int r  = s / 20;
        int sg = s - r * 20;                  // 0..9 -> h, 10..19 -> m
        long row = rowBase + r;
        if (row >= NATOMS) continue;
        unsigned lds = ldsAoff + (unsigned)r * (KPAD * 2) + (unsigned)sg * 16;
        if (sg < 10) {
            unsigned voff = (unsigned)row * (FP * 2) + (unsigned)sg * 16;
            asm volatile("global_load_async_to_lds_b128 %0, %1, %2"
                         :: "v"(lds), "v"(voff), "s"((const void*)h) : "memory");
        } else {
            unsigned voff = (unsigned)row * (FP * 2) + (unsigned)(sg - 10) * 16;
            asm volatile("global_load_async_to_lds_b128 %0, %1, %2"
                         :: "v"(lds), "v"(voff), "s"((const void*)m) : "memory");
        }
    }

    // Stage B = W^T (padded, fragment-major). Padded k -> original k:
    // k<80 -> h col k (valid k<75); k>=80 -> m col k-80 (valid k<155).
    for (int s = tid; s < 5 * 5 * 32 * 16; s += 128) {
        int i  = s & 15;
        int ln = (s >> 4) & 31;
        int t  = s >> 9;            // 0..24
        int kc = t % 5, nt = t / 5;
        int j  = nt * 16 + (ln & 15);
        int k  = kc * 32 + ((ln & 16) ? 16 : 0) + i;
        int ko = (k < FP) ? k : (k - (FP - FDIM));
        bool ok = (j < FDIM) && ((k < FP) ? (k < FDIM) : (k < FP + FDIM));
        ldsB[s] = ok ? f2bf(W[j * (2 * FDIM) + ko]) : (ushort)0;
    }

    asm volatile("s_wait_asynccnt 0x0" ::: "memory");
    __syncthreads();

    v8f acc[5] = {};
    for (int kc = 0; kc < 5; ++kc) {
        // A fragment: lanes 0-15 -> K {0..15}, lanes 16-31 -> K {16..31}
        ABfrag a;
        const ushort* pa = &ldsA[(wave * 16 + rl) * KPAD + kc * 32 + hi * 16];
        a.q[0] = *(const v4u*)pa;
        a.q[1] = *(const v4u*)(pa + 8);
        for (int nt = 0; nt < 5; ++nt) {
            ABfrag bfr;
            const ushort* pb = &ldsB[(((nt * 5 + kc) * 32) + lane) * 16];
            bfr.q[0] = *(const v4u*)pb;
            bfr.q[1] = *(const v4u*)(pb + 8);
            acc[nt] = __builtin_amdgcn_wmma_f32_16x16x32_bf16(
                false, a.bf, false, bfr.bf, (short)0, acc[nt], false, false);
        }
    }

    // D layout: VGPR v -> row v + 8*hi, col = lane&15. Bias+ReLU, bf16 store,
    // pad cols 75..79 written as zero (required by the packed scatter).
    long r0 = rowBase + wave * 16 + hi * 8;
    for (int nt = 0; nt < 5; ++nt) {
        int col = nt * 16 + rl;
        float bj = (col < FDIM) ? bias[col] : 0.0f;
        for (int v = 0; v < 8; ++v) {
            long row = r0 + v;
            if (row < NATOMS) {
                float val = (col < FDIM) ? fmaxf(acc[nt][v] + bj, 0.0f) : 0.0f;
                out[row * FP + col] = f2bf(val);
            }
        }
    }
}

// --------------------- fused readout: out[mol] = NN_b + sum <h[a], NN_W>
__global__ void init_out(float* __restrict__ out, const float* __restrict__ NN_b) {
    int i = blockIdx.x * blockDim.x + threadIdx.x;
    if (i < NMOLS) out[i] = NN_b[0];
}

__global__ void atom_reduce(const ushort* __restrict__ h,
                            const float* __restrict__ NN_W,
                            const int*   __restrict__ mol_ids,
                            float*       __restrict__ out) {
    int a    = blockIdx.x * (blockDim.x >> 5) + (threadIdx.x >> 5);
    int lane = threadIdx.x & 31;
    if (a >= NATOMS) return;
    float s = 0.0f;
    for (int j = lane; j < FDIM; j += 32)
        s += bf2f(h[(long)a * FP + j]) * NN_W[j];
    for (int off = 16; off > 0; off >>= 1)
        s += __shfl_xor(s, off, 32);
    if (lane == 0) atomic_fadd(&out[mol_ids[a]], s);
}

// ---------------------------------------------------------------- launcher
extern "C" void kernel_launch(void* const* d_in, const int* in_sizes, int n_in,
                              void* d_out, int out_size, void* d_ws, size_t ws_size,
                              hipStream_t stream) {
    const float* atom_feats = (const float*)d_in[0];
    const float* U_W        = (const float*)d_in[1];
    const float* U_b        = (const float*)d_in[2];
    const float* NN_W       = (const float*)d_in[3];
    const float* NN_b       = (const float*)d_in[4];
    const int*   edge_src   = (const int*)d_in[5];
    const int*   edge_dst   = (const int*)d_in[6];
    const int*   mol_ids    = (const int*)d_in[7];
    float*       out        = (float*)d_out;

    const size_t nh = (size_t)NATOMS * FP;    // bf16 elements per buffer
    ushort* mbuf = (ushort*)d_ws;
    ushort* hA   = mbuf + nh;
    ushort* hB   = hA + nh;

    // atom_feats (fp32) -> bf16 padded rows
    cvt_feats<<<(unsigned)((nh + 255) / 256), 256, 0, stream>>>(atom_feats, hA);

    const long ne = (long)NEDGES * (FP / 2);
    const ushort* hsrc = hA;
    ushort*       hdst = hB;

    for (int d = 0; d < DEPTH; ++d) {
        zero_u32<<<(unsigned)((nh / 2 + 255) / 256), 256, 0, stream>>>(
            (unsigned*)mbuf, (long)(nh / 2));
        edge_scatter<<<(unsigned)((ne + 255) / 256), 256, 0, stream>>>(
            hsrc, edge_src, edge_dst, mbuf);
        mpnn_gemm<<<(unsigned)((NATOMS + 63) / 64), 128, 0, stream>>>(
            hsrc, mbuf, U_W, U_b, hdst);
        const ushort* t = hsrc; hsrc = hdst; hdst = (ushort*)t;
    }

    init_out<<<(NMOLS + 255) / 256, 256, 0, stream>>>(out, NN_b);
    atom_reduce<<<(NATOMS * 32 + 255) / 256, 256, 0, stream>>>(
        hsrc, NN_W, mol_ids, out);
}
